// EncoderStack_64647847739883
// MI455X (gfx1250) — compile-verified, matrix operand order checked by reading, not run on hardware
//
#include <hip/hip_runtime.h>
#include <hip/hip_bf16.h>
#include <stdint.h>

// Problem dims (fixed by the reference)
#define BSZ 2
#define T   2048
#define D   1024
#define H   16
#define DKV 64
#define FF  4096
#define NTOK 4096          // BSZ*T
#define NQKV 3072          // 3*H*DKV

typedef __attribute__((ext_vector_type(16))) __bf16 bf16x16;
typedef __attribute__((ext_vector_type(8)))  float  f32x8;

union FragCast { bf16x16 v; uint4 u[2]; };

#if defined(__AMDGCN__) && __has_builtin(__builtin_amdgcn_global_load_async_to_lds_b128)
#define HAVE_ASYNC_LDS 1
#else
#define HAVE_ASYNC_LDS 0
#endif

typedef int v4i_ __attribute__((vector_size(16)));
typedef __attribute__((address_space(1))) v4i_* gv4i_p;
typedef __attribute__((address_space(3))) v4i_* lv4i_p;

__device__ __forceinline__ f32x8 wmma_bf16(bf16x16 a, bf16x16 b, f32x8 c) {
  return __builtin_amdgcn_wmma_f32_16x16x32_bf16(false, a, false, b, (short)0, c,
                                                 false, false);
}

__device__ __forceinline__ f32x8 zero8() {
  f32x8 z;
  for (int e = 0; e < 8; ++e) z[e] = 0.0f;
  return z;
}

// Load a 16x32 A-fragment (or B^T-fragment) from row-major bf16 memory
// (global or LDS). Lane l holds row (l&15); K = 16*(j>>3) + 8*(l>>4) + (j&7).
// Each 8-half group is contiguous -> two 16B loads.
__device__ __forceinline__ bf16x16 load_frag(const __bf16* base, int row0,
                                             int ld, int k0) {
  int lane = threadIdx.x & 31;
  int m = lane & 15;
  int kh = lane >> 4;
  const __bf16* p = base + (size_t)(row0 + m) * ld + k0 + 8 * kh;
  FragCast f;
  f.u[0] = *(const uint4*)(const void*)(p);
  f.u[1] = *(const uint4*)(const void*)(p + 16);
  return f.v;
}

// Stage one 32-half (64B) row segment global -> LDS.
__device__ __forceinline__ void stage_row(const __bf16* g, __bf16* s) {
#if HAVE_ASYNC_LDS
#pragma unroll
  for (int c = 0; c < 4; ++c) {
    gv4i_p gp = (gv4i_p)(v4i_*)(void*)(const_cast<__bf16*>(g) + 8 * c);
    lv4i_p lp = (lv4i_p)(v4i_*)(void*)(s + 8 * c);
    __builtin_amdgcn_global_load_async_to_lds_b128(gp, lp, 0, 0);
  }
#else
  uint4 u[4];
#pragma unroll
  for (int c = 0; c < 4; ++c) u[c] = *(const uint4*)(const void*)(g + 8 * c);
#pragma unroll
  for (int c = 0; c < 4; ++c) *(uint4*)(void*)(s + 8 * c) = u[c];
#endif
}

__device__ __forceinline__ void wait_stage() {
#if HAVE_ASYNC_LDS
#if __has_builtin(__builtin_amdgcn_s_wait_asynccnt)
  __builtin_amdgcn_s_wait_asynccnt(0);
#else
  asm volatile("s_wait_asynccnt 0x0" ::: "memory");
#endif
#endif
}

// ---------------------------------------------------------------------------
// GEMM: C(MxN) = A(MxK bf16 rowmajor) * Bt(NxK bf16 rowmajor)^T
// MODE 0: store bf16 | 1: +bias,ReLU,bf16 | 2: +resid f32 | 3: +bias+resid f32
// 256 thr = 8 waves (2x4). Block tile 128x128, BK=32, wave tile 64x32.
// Double-buffered LDS staging (async-to-LDS on CDNA5).
// ---------------------------------------------------------------------------
template <int MODE>
__global__ __launch_bounds__(256) void gemm_bf16_tn(
    const __bf16* __restrict__ A, const __bf16* __restrict__ Bt, int M, int N,
    int K, __bf16* __restrict__ outB, float* __restrict__ outF,
    const float* __restrict__ bias, const float* __restrict__ resid) {
  __shared__ __align__(16) __bf16 smA[2][128 * 32];
  __shared__ __align__(16) __bf16 smB[2][128 * 32];

  int t = threadIdx.x;
  int wave = t >> 5;
  int lane = t & 31;
  int browg = blockIdx.y * 128;
  int bcolg = blockIdx.x * 128;

  // staging assignment: threads 0..127 stage A rows, 128..255 stage Bt rows
  int rrow = t & 127;
  const __bf16* gsrc = (t < 128) ? A + (size_t)(browg + rrow) * K
                                 : Bt + (size_t)(bcolg + rrow) * K;
  __bf16* sd[2] = {((t < 128) ? smA[0] : smB[0]) + rrow * 32,
                   ((t < 128) ? smA[1] : smB[1]) + rrow * 32};

  f32x8 acc[4][2];
  for (int i = 0; i < 4; ++i)
    for (int j = 0; j < 2; ++j) acc[i][j] = zero8();

  // prologue: stage tile 0
  stage_row(gsrc, sd[0]);
  wait_stage();
  __syncthreads();

  int row0w = (wave >> 2) * 64;  // 0 / 64
  int col0w = (wave & 3) * 32;   // 0..96
  int cur = 0;

  for (int kk = 32; kk <= K; kk += 32) {
    if (kk < K) stage_row(gsrc + kk, sd[cur ^ 1]);

    const __bf16* sa = smA[cur];
    const __bf16* sb = smB[cur];
    bf16x16 af[4], bfv[2];
#pragma unroll
    for (int i = 0; i < 4; ++i) af[i] = load_frag(sa, row0w + 16 * i, 32, 0);
#pragma unroll
    for (int j = 0; j < 2; ++j) bfv[j] = load_frag(sb, col0w + 16 * j, 32, 0);
#pragma unroll
    for (int i = 0; i < 4; ++i)
#pragma unroll
      for (int j = 0; j < 2; ++j)
        acc[i][j] = wmma_bf16(af[i], bfv[j], acc[i][j]);

    if (kk < K) wait_stage();
    __syncthreads();
    cur ^= 1;
  }

  int mh = (lane >> 4) * 8;
  int nn = lane & 15;
#pragma unroll
  for (int i = 0; i < 4; ++i)
#pragma unroll
    for (int j = 0; j < 2; ++j) {
      int gc = bcolg + col0w + 16 * j + nn;
#pragma unroll
      for (int r = 0; r < 8; ++r) {
        int gr = browg + row0w + 16 * i + mh + r;
        size_t idx = (size_t)gr * N + gc;
        float v = acc[i][j][r];
        if (MODE == 1) {
          v += bias[gc];
          v = v > 0.0f ? v : 0.0f;
        }
        if (MODE == 3) v += bias[gc];
        if (MODE == 0 || MODE == 1)
          outB[idx] = (__bf16)v;
        else
          outF[idx] = v + resid[idx];
      }
    }
}

// ---------------------------------------------------------------------------
// Attention pass A: column softmax stats (softmax is over the QUERY axis).
// Grid: (B*H)*(T/128); each wave owns 16 key columns, loops all q-tiles.
// ---------------------------------------------------------------------------
__global__ __launch_bounds__(256) void attn_colstats(
    const __bf16* __restrict__ qb, const __bf16* __restrict__ kb,
    float* __restrict__ cmax, float* __restrict__ csum) {
  const int sblocks = T / 128;
  int bh = blockIdx.x / sblocks;
  int sblk = blockIdx.x % sblocks;
  int wave = threadIdx.x >> 5;
  int lane = threadIdx.x & 31;
  int s0 = sblk * 128 + wave * 16;

  const __bf16* qh = qb + (size_t)bh * T * DKV;
  const __bf16* kh = kb + (size_t)bh * T * DKV;

  bf16x16 kf0 = load_frag(kh, s0, DKV, 0);
  bf16x16 kf1 = load_frag(kh, s0, DKV, 32);

  float m = -3.0e38f, l = 0.0f;
  for (int qt = 0; qt < T; qt += 16) {
    bf16x16 a0 = load_frag(qh, qt, DKV, 0);
    bf16x16 a1 = load_frag(qh, qt, DKV, 32);
    f32x8 c = zero8();
    c = wmma_bf16(a0, kf0, c);
    c = wmma_bf16(a1, kf1, c);
    float sc[8], tmax = -3.0e38f;
#pragma unroll
    for (int r = 0; r < 8; ++r) {
      sc[r] = c[r] * 0.125f;  // 1/sqrt(64)
      tmax = fmaxf(tmax, sc[r]);
    }
    // lanes l and l+16 hold the same key column (different q rows): fuse
    tmax = fmaxf(tmax, __shfl_xor(tmax, 16, 32));
    float mn = fmaxf(m, tmax);
    float p = 0.0f;
#pragma unroll
    for (int r = 0; r < 8; ++r) p += __expf(sc[r] - mn);
    p += __shfl_xor(p, 16, 32);
    l = l * __expf(m - mn) + p;
    m = mn;
  }
  if (lane < 16) {
    cmax[(size_t)bh * T + s0 + lane] = m;
    csum[(size_t)bh * T + s0 + lane] = l;
  }
}

// ---------------------------------------------------------------------------
// Attention pass B: recompute scores, p = exp(s/8 - m)/sum, o += p @ v.
// p: C-layout -> LDS -> A-fragment. vT supplies B^T rows directly.
// ---------------------------------------------------------------------------
__global__ __launch_bounds__(256) void attn_out(
    const __bf16* __restrict__ qb, const __bf16* __restrict__ kb,
    const __bf16* __restrict__ vtb, const float* __restrict__ cmax,
    const float* __restrict__ csum, __bf16* __restrict__ oc) {
  __shared__ __align__(16) __bf16 pbuf[8][16 * 40];  // per-wave 16x32 (+pad)
  const int qblocks = T / 128;
  int bh = blockIdx.x / qblocks;
  int qblk = blockIdx.x % qblocks;
  int wave = threadIdx.x >> 5;
  int lane = threadIdx.x & 31;
  int q0 = qblk * 128 + wave * 16;

  const __bf16* qh = qb + (size_t)bh * T * DKV;
  const __bf16* kh = kb + (size_t)bh * T * DKV;
  const __bf16* vh = vtb + (size_t)bh * DKV * T;

  bf16x16 aq0 = load_frag(qh, q0, DKV, 0);
  bf16x16 aq1 = load_frag(qh, q0, DKV, 32);

  f32x8 oacc[4];
  for (int n = 0; n < 4; ++n) oacc[n] = zero8();

  int nn = lane & 15;
  int mh = (lane >> 4) * 8;
  __bf16* myp = &pbuf[wave][0];

  for (int s0 = 0; s0 < T; s0 += 32) {
    f32x8 c0 = zero8(), c1 = zero8();
    bf16x16 k00 = load_frag(kh, s0, DKV, 0);
    bf16x16 k01 = load_frag(kh, s0, DKV, 32);
    c0 = wmma_bf16(aq0, k00, c0);
    c0 = wmma_bf16(aq1, k01, c0);
    bf16x16 k10 = load_frag(kh, s0 + 16, DKV, 0);
    bf16x16 k11 = load_frag(kh, s0 + 16, DKV, 32);
    c1 = wmma_bf16(aq0, k10, c1);
    c1 = wmma_bf16(aq1, k11, c1);

    size_t sb = (size_t)bh * T;
    float m0 = cmax[sb + s0 + nn], r0 = 1.0f / csum[sb + s0 + nn];
    float m1 = cmax[sb + s0 + 16 + nn], r1 = 1.0f / csum[sb + s0 + 16 + nn];
#pragma unroll
    for (int r = 0; r < 8; ++r) {
      myp[(mh + r) * 40 + nn] = (__bf16)(__expf(c0[r] * 0.125f - m0) * r0);
      myp[(mh + r) * 40 + nn + 16] = (__bf16)(__expf(c1[r] * 0.125f - m1) * r1);
    }
    __syncthreads();
    bf16x16 pf = load_frag(myp, 0, 40, 0);  // 16 q rows x K=32 (s)
#pragma unroll
    for (int n = 0; n < 4; ++n) {
      bf16x16 bv = load_frag(vh, n * 16, T, s0);  // vT rows = v columns
      oacc[n] = wmma_bf16(pf, bv, oacc[n]);
    }
    __syncthreads();
  }

  int b = bh / H, h = bh % H;
#pragma unroll
  for (int n = 0; n < 4; ++n)
#pragma unroll
    for (int r = 0; r < 8; ++r) {
      int gt = q0 + mh + r;
      int col = h * DKV + n * 16 + nn;
      oc[((size_t)b * T + gt) * D + col] = (__bf16)oacc[n][r];
    }
}

// ---------------------------------------------------------------------------
// Row mean/std norm (ddof=1). One block per row of D=1024.
// ---------------------------------------------------------------------------
__global__ __launch_bounds__(256) void rownorm(const float* __restrict__ y,
                                               float* __restrict__ outF,
                                               __bf16* __restrict__ outB) {
  __shared__ float s1[256], s2[256];
  int row = blockIdx.x, tid = threadIdx.x;
  const float* pr = y + (size_t)row * D;
  float v[4], s = 0.0f, q = 0.0f;
#pragma unroll
  for (int j = 0; j < 4; ++j) {
    v[j] = pr[tid + 256 * j];
    s += v[j];
    q += v[j] * v[j];
  }
  s1[tid] = s;
  s2[tid] = q;
  __syncthreads();
  for (int o = 128; o > 0; o >>= 1) {
    if (tid < o) {
      s1[tid] += s1[tid + o];
      s2[tid] += s2[tid + o];
    }
    __syncthreads();
  }
  float mean = s1[0] * (1.0f / D);
  float var = (s2[0] - s1[0] * s1[0] * (1.0f / D)) * (1.0f / (D - 1));
  float rstd = rsqrtf(var);
  float* po = outF + (size_t)row * D;
#pragma unroll
  for (int j = 0; j < 4; ++j) {
    float o = (v[j] - mean) * rstd;
    po[tid + 256 * j] = o;
    if (outB) outB[(size_t)row * D + tid + 256 * j] = (__bf16)o;
  }
}

// ------------------------- conversion / layout kernels ---------------------
__global__ void cvt_f32_bf16(const float* __restrict__ in,
                             __bf16* __restrict__ out, int n) {
  int i = blockIdx.x * 256 + threadIdx.x;
  if (i < n) out[i] = (__bf16)in[i];
}

// WqkvT[n][d], n = which*1024 + h*64 + k  (rows = GEMM columns, K-contig)
__global__ void pack_wqkvT(const float* __restrict__ Wq,
                           const float* __restrict__ Wk,
                           const float* __restrict__ Wv,
                           __bf16* __restrict__ out) {
  int idx = blockIdx.x * 256 + threadIdx.x;
  if (idx >= NQKV * D) return;
  int d = idx & (D - 1);
  int n = idx >> 10;
  int which = n >> 10;
  int rem = n & 1023;
  int h = rem >> 6;
  int kk = rem & 63;
  const float* src = which == 0 ? Wq : (which == 1 ? Wk : Wv);
  out[idx] = (__bf16)src[((size_t)h * D + d) * DKV + kk];
}

// out (C x R) = transpose of in (R x C), bf16
__global__ void packT(const float* __restrict__ in, __bf16* __restrict__ out,
                      int R, int C) {
  int idx = blockIdx.x * 256 + threadIdx.x;
  if (idx >= R * C) return;
  int orow = idx / R;  // = column of in
  int ocol = idx % R;  // = row of in
  out[idx] = (__bf16)in[(size_t)ocol * C + orow];
}

// qkv (NTOK x 3072) -> q,k (B,H,T,64) and vT (B,H,64,T)
__global__ void split_qkv(const __bf16* __restrict__ qkv,
                          __bf16* __restrict__ q, __bf16* __restrict__ k,
                          __bf16* __restrict__ vT) {
  int idx = blockIdx.x * 256 + threadIdx.x;
  if (idx >= NTOK * NQKV) return;
  int n = idx % NQKV;
  int tok = idx / NQKV;
  int which = n >> 10, rem = n & 1023;
  int h = rem >> 6, kk = rem & 63;
  int b = tok >> 11, t = tok & 2047;
  __bf16 val = qkv[idx];
  size_t bhb = (size_t)(b * H + h);
  if (which == 0)
    q[(bhb * T + t) * DKV + kk] = val;
  else if (which == 1)
    k[(bhb * T + t) * DKV + kk] = val;
  else
    vT[(bhb * DKV + kk) * T + t] = val;
}

// ---------------------------------------------------------------------------
extern "C" void kernel_launch(void* const* d_in, const int* in_sizes, int n_in,
                              void* d_out, int out_size, void* d_ws,
                              size_t ws_size, hipStream_t stream) {
  const float* x = (const float*)d_in[0];
  const float* Wq = (const float*)d_in[1];
  const float* Wk = (const float*)d_in[2];
  const float* Wv = (const float*)d_in[3];
  const float* Wo = (const float*)d_in[4];
  const float* W1 = (const float*)d_in[5];
  const float* b1 = (const float*)d_in[6];
  const float* W2 = (const float*)d_in[7];
  const float* b2 = (const float*)d_in[8];

  char* ws = (char*)d_ws;
  const size_t OFF_XB = 0;             //  8 MiB
  const size_t OFF_WQKVT = 8388608;    //  6 MiB
  const size_t OFF_WOT = 14680064;     //  2 MiB
  const size_t OFF_W1T = 16777216;     //  8 MiB
  const size_t OFF_W2T = 25165824;     //  8 MiB
  const size_t OFF_QKV = 33554432;     // 24 MiB (reused as y2 later)
  const size_t OFF_QB = 58720256;      //  8 MiB
  const size_t OFF_KB = 67108864;      //  8 MiB
  const size_t OFF_VT = 75497472;      //  8 MiB
  const size_t OFF_CMAX = 83886080;    // 256 KiB
  const size_t OFF_CSUM = 84148224;    // 256 KiB
  const size_t OFF_OC = 84410368;      //  8 MiB
  const size_t OFF_Y1 = 92798976;      // 16 MiB f32
  const size_t OFF_OUT1F = 109576192;  // 16 MiB f32
  const size_t OFF_OUT1B = 126353408;  //  8 MiB
  const size_t OFF_HB = 134742016;     // 32 MiB

  __bf16* xb = (__bf16*)(ws + OFF_XB);
  __bf16* wqkvT = (__bf16*)(ws + OFF_WQKVT);
  __bf16* woT = (__bf16*)(ws + OFF_WOT);
  __bf16* w1T = (__bf16*)(ws + OFF_W1T);
  __bf16* w2T = (__bf16*)(ws + OFF_W2T);
  __bf16* qkvb = (__bf16*)(ws + OFF_QKV);
  __bf16* qb = (__bf16*)(ws + OFF_QB);
  __bf16* kb = (__bf16*)(ws + OFF_KB);
  __bf16* vtb = (__bf16*)(ws + OFF_VT);
  float* cmax = (float*)(ws + OFF_CMAX);
  float* csum = (float*)(ws + OFF_CSUM);
  __bf16* ocb = (__bf16*)(ws + OFF_OC);
  float* y1 = (float*)(ws + OFF_Y1);
  float* out1f = (float*)(ws + OFF_OUT1F);
  __bf16* out1b = (__bf16*)(ws + OFF_OUT1B);
  __bf16* hb = (__bf16*)(ws + OFF_HB);
  float* y2 = (float*)(ws + OFF_QKV);  // alias: qkv dead after split

  // 1) convert x; pack transposed bf16 weights
  cvt_f32_bf16<<<(NTOK * D + 255) / 256, 256, 0, stream>>>(x, xb, NTOK * D);
  pack_wqkvT<<<(NQKV * D + 255) / 256, 256, 0, stream>>>(Wq, Wk, Wv, wqkvT);
  packT<<<(D * D + 255) / 256, 256, 0, stream>>>(Wo, woT, D, D);
  packT<<<(D * FF + 255) / 256, 256, 0, stream>>>(W1, w1T, D, FF);
  packT<<<(FF * D + 255) / 256, 256, 0, stream>>>(W2, w2T, FF, D);

  // 2) fused QKV projection GEMM: (4096x1024)x(1024x3072)
  gemm_bf16_tn<0><<<dim3(NQKV / 128, NTOK / 128), 256, 0, stream>>>(
      xb, wqkvT, NTOK, NQKV, D, qkvb, nullptr, nullptr, nullptr);
  split_qkv<<<(NTOK * NQKV + 255) / 256, 256, 0, stream>>>(qkvb, qb, kb, vtb);

  // 3) attention (column softmax, two-pass flash-style over q)
  attn_colstats<<<(BSZ * H) * (T / 128), 256, 0, stream>>>(qb, kb, cmax, csum);
  attn_out<<<(BSZ * H) * (T / 128), 256, 0, stream>>>(qb, kb, vtb, cmax, csum,
                                                      ocb);

  // 4) output projection + residual, then norm
  gemm_bf16_tn<2><<<dim3(D / 128, NTOK / 128), 256, 0, stream>>>(
      ocb, woT, NTOK, D, D, nullptr, y1, nullptr, x);
  rownorm<<<NTOK, 256, 0, stream>>>(y1, out1f, out1b);

  // 5) FFN: relu(out1 @ W1 + b1) @ W2 + b2 + out1, then norm -> d_out
  gemm_bf16_tn<1><<<dim3(FF / 128, NTOK / 128), 256, 0, stream>>>(
      out1b, w1T, NTOK, FF, D, hb, nullptr, b1, nullptr);
  gemm_bf16_tn<3><<<dim3(D / 128, NTOK / 128), 256, 0, stream>>>(
      hb, w2T, NTOK, D, FF, nullptr, y2, b2, out1f);
  rownorm<<<NTOK, 256, 0, stream>>>(y2, (float*)d_out, nullptr);

  (void)in_sizes;
  (void)n_in;
  (void)out_size;
  (void)ws_size;
}
